// SampleRadianceField_47141561041210
// MI455X (gfx1250) — matrix operand order repaired and checked by reading, not compiled, not run on hardware
//
#include <hip/hip_runtime.h>
#include <hip/hip_bf16.h>

typedef __attribute__((ext_vector_type(16))) _Float16 v16h;
typedef __attribute__((ext_vector_type(8)))  float    v8f;

#define LVLS   16
#define TBL    1024
#define FEATF  2
#define NTAB   (LVLS * TBL * FEATF)   // 32768 floats = 128 KB
#define IMG_W  1920
#define IMG_H  1080

// fast stable softplus: v_exp_f32 + v_log_f32 (TRANS32, co-executes with VALU)
__device__ __forceinline__ float fast_softplus(float x)
{
    float e = __expf(-fabsf(x));
    return fmaxf(x, 0.0f) + __logf(1.0f + e);
}

// -------------------------------------------------------------------------
// Kernel A: camera projection + bilinear image sample (color output)
// -------------------------------------------------------------------------
__global__ __launch_bounds__(256)
void color_kernel(const float* __restrict__ xyzs,
                  const float* __restrict__ img,
                  const float* __restrict__ Kmat,
                  const float* __restrict__ Twc,
                  float* __restrict__ color,
                  int N)
{
    int p = blockIdx.x * blockDim.x + threadIdx.x;
    if (p >= N) return;

    // R_cw = R_wc^T ; t_cw = -R_cw @ t_wc   (Twc is 4x4 row-major, uniform -> scalar loads)
    float R[9], t[3];
    #pragma unroll
    for (int i = 0; i < 3; ++i) {
        #pragma unroll
        for (int j = 0; j < 3; ++j) R[i*3 + j] = Twc[j*4 + i];
    }
    #pragma unroll
    for (int i = 0; i < 3; ++i)
        t[i] = -(R[i*3+0]*Twc[0*4+3] + R[i*3+1]*Twc[1*4+3] + R[i*3+2]*Twc[2*4+3]);

    float Xw = xyzs[3*p+0], Yw = xyzs[3*p+1], Zw = xyzs[3*p+2];
    float xc = R[0]*Xw + R[1]*Yw + R[2]*Zw + t[0];
    float yc = R[3]*Xw + R[4]*Yw + R[5]*Zw + t[1];
    float zc = R[6]*Xw + R[7]*Yw + R[8]*Zw + t[2];
    if (fabsf(zc) < 0.001f) zc = 0.001f;
    float xz = xc / zc, yz = yc / zc;

    float px = Kmat[0]*xz + Kmat[1]*yz + Kmat[2];
    float py = Kmat[3]*xz + Kmat[4]*yz + Kmat[5];

    px = fminf(fmaxf(px, 0.0f), (float)(IMG_W - 2));
    py = fminf(fmaxf(py, 0.0f), (float)(IMG_H - 2));
    float fx0 = floorf(px), fy0 = floorf(py);
    float fx = px - fx0, fy = py - fy0;
    int x0 = (int)fx0, y0 = (int)fy0;

    const float* c00 = img + (size_t)(y0 * IMG_W + x0) * 3;
    const float* c10 = c00 + 3;
    const float* c01 = c00 + (size_t)IMG_W * 3;
    const float* c11 = c01 + 3;

    #pragma unroll
    for (int c = 0; c < 3; ++c) {
        float a = fx * c10[c] + (1.0f - fx) * c00[c];
        float b = fx * c11[c] + (1.0f - fx) * c01[c];
        color[3*p + c] = fy * b + (1.0f - fy) * a;
    }
}

// -------------------------------------------------------------------------
// Kernel B: hash-grid encode (tables in LDS) + WMMA MLP -> sigma
//   One wave = 16 points per tile. Lane L builds the A-fragment slice that
//   the 16x16x32 f16 WMMA layout assigns to it:
//     lane<16 : K 0..7 (elts 0..7)  and K 16..23 (elts 8..15)
//     lane>=16: K 8..15(elts 0..7)  and K 24..31 (elts 8..15)
//   with K = 2*level + f  => lane computes levels 4*hi+8g .. +3  (g=0,1).
// -------------------------------------------------------------------------
__global__ __launch_bounds__(256)
void sigma_kernel(const float* __restrict__ xyzs,
                  const float* __restrict__ tables,
                  const float* __restrict__ W1,
                  const float* __restrict__ W2,
                  float* __restrict__ sigma,
                  int N)
{
    __shared__ __align__(16) float s_tab[NTAB];   // 128 KB of 320 KB WGP LDS

    const int tid = threadIdx.x;
    // cooperative 128 KB preload, float4 granularity (global b128 -> ds_store_b128)
    for (int i = tid; i < NTAB / 4; i += 256)
        ((float4*)s_tab)[i] = ((const float4*)tables)[i];
    __syncthreads();

    const int lane = tid & 31;
    const int wave = tid >> 5;
    const int hi   = lane >> 4;       // which half of the wave
    const int col  = lane & 15;
    const int gw   = blockIdx.x * 8 + wave;
    const int nwaves = gridDim.x * 8;
    const int stride = nwaves * 16;

    // ---- W1 B-fragments (32x16 K-major per tile), loaded once per wave ----
    v16h b0, b1, b2, b3;
    #pragma unroll
    for (int i = 0; i < 16; ++i) {
        int k = (i & 7) + 8 * hi + 16 * (i >> 3);   // B elt i holds K = 8*hi + i (i<8), 16 + 8*hi + (i-8) (i>=8)
        b0[i] = (_Float16)W1[k*64 + col +  0];
        b1[i] = (_Float16)W1[k*64 + col + 16];
        b2[i] = (_Float16)W1[k*64 + col + 32];
        b3[i] = (_Float16)W1[k*64 + col + 48];
    }
    // W2 column 0 element this lane needs per N-tile
    float w2f[4];
    #pragma unroll
    for (int tt = 0; tt < 4; ++tt) w2f[tt] = W2[(16*tt + col) * 16 + 0];

    // level scales: scale_l = 16 * 2^(8l/15) - 1
    float lscale[2][4];
    #pragma unroll
    for (int g = 0; g < 2; ++g)
        #pragma unroll
        for (int j = 0; j < 4; ++j) {
            int lv = 4*hi + 8*g + j;
            lscale[g][j] = 16.0f * exp2f((8.0f/15.0f) * (float)lv) - 1.0f;
        }

    const unsigned P1 = 2654435761u, P2 = 805459861u;

    for (int base = gw * 16; base < N; base += stride) {
        int p = base + col;           // both halves hold point base+col
        if (p > N - 1) p = N - 1;     // clamp: keep EXEC all-1s for WMMA

        // prefetch next tile's xyz into cache (global_prefetch_b8)
        {
            int pn = base + stride + col;
            if (pn < N) __builtin_prefetch(&xyzs[3 * pn], 0, 3);
        }

        float x01 = (xyzs[3*p+0] + 1.0f) * 0.5f;
        float y01 = (xyzs[3*p+1] + 1.0f) * 0.5f;
        float z01 = (xyzs[3*p+2] + 1.0f) * 0.5f;

        v16h a;
        #pragma unroll
        for (int g = 0; g < 2; ++g) {
            #pragma unroll
            for (int j = 0; j < 4; ++j) {
                int   level = 4*hi + 8*g + j;
                float s = lscale[g][j];
                float pxf = x01 * s + 0.5f;
                float pyf = y01 * s + 0.5f;
                float pzf = z01 * s + 0.5f;
                float flx = floorf(pxf), fly = floorf(pyf), flz = floorf(pzf);
                float frx = pxf - flx, fry = pyf - fly, frz = pzf - flz;
                unsigned cx = (unsigned)flx, cy = (unsigned)fly, cz = (unsigned)flz;

                float acc0 = 0.0f, acc1 = 0.0f;
                const float2* tl = ((const float2*)s_tab) + level * TBL;
                #pragma unroll
                for (int corner = 0; corner < 8; ++corner) {
                    unsigned ox = corner & 1, oy = (corner >> 1) & 1, oz = corner >> 2;
                    unsigned idx = ((cx + ox) ^ ((cy + oy) * P1) ^ ((cz + oz) * P2)) & (TBL - 1);
                    float w = (ox ? frx : 1.0f - frx) *
                              (oy ? fry : 1.0f - fry) *
                              (oz ? frz : 1.0f - frz);
                    float2 f = tl[idx];                 // ds_load_b64 from LDS
                    acc0 += w * f.x;
                    acc1 += w * f.y;
                }
                a[8*g + 2*j + 0] = (_Float16)acc0;
                a[8*g + 2*j + 1] = (_Float16)acc1;
            }
        }

        // ---- hidden = feats(16x32) @ W1(32x64): 4 x v_wmma_f32_16x16x32_f16 ----
        v8f c0 = {}, c1 = {}, c2 = {}, c3 = {};
        c0 = __builtin_amdgcn_wmma_f32_16x16x32_f16(false, a, false, b0, (short)0, c0, false, false);
        c1 = __builtin_amdgcn_wmma_f32_16x16x32_f16(false, a, false, b1, (short)0, c1, false, false);
        c2 = __builtin_amdgcn_wmma_f32_16x16x32_f16(false, a, false, b2, (short)0, c2, false, false);
        c3 = __builtin_amdgcn_wmma_f32_16x16x32_f16(false, a, false, b3, (short)0, c3, false, false);

        // ---- sigma = exp( softplus(hidden) . W2[:,0] ) ----
        // C layout: VGPR r, lane: hidden[m = r + 8*hi][n = 16*t + col]
        float part[8];
        #pragma unroll
        for (int r = 0; r < 8; ++r) part[r] = 0.0f;
        #pragma unroll
        for (int tt = 0; tt < 4; ++tt) {
            v8f ct = (tt == 0) ? c0 : (tt == 1) ? c1 : (tt == 2) ? c2 : c3;
            float w2 = w2f[tt];
            #pragma unroll
            for (int r = 0; r < 8; ++r)
                part[r] += fast_softplus(ct[r]) * w2;
        }
        // reduce over the 16 lanes of each half (halves stay independent)
        #pragma unroll
        for (int m = 1; m < 16; m <<= 1)
            #pragma unroll
            for (int r = 0; r < 8; ++r)
                part[r] += __shfl_xor(part[r], m, 32);

        if (col == 0) {
            #pragma unroll
            for (int r = 0; r < 8; ++r) {
                int pp = base + r + 8*hi;
                if (pp < N) sigma[pp] = __expf(part[r]);
            }
        }
    }
}

// -------------------------------------------------------------------------
extern "C" void kernel_launch(void* const* d_in, const int* in_sizes, int n_in,
                              void* d_out, int out_size, void* d_ws, size_t ws_size,
                              hipStream_t stream)
{
    const float* xyzs = (const float*)d_in[0];
    // d_in[1] = dirs (unused by reference)
    const float* img  = (const float*)d_in[2];
    const float* Kmat = (const float*)d_in[3];
    const float* Twc  = (const float*)d_in[4];
    const float* tab  = (const float*)d_in[5];
    const float* W1   = (const float*)d_in[6];
    const float* W2   = (const float*)d_in[7];

    const int N = in_sizes[0] / 3;
    float* color = (float*)d_out;           // [N,3]
    float* sigma = color + (size_t)3 * N;   // [N]

    color_kernel<<<(N + 255) / 256, 256, 0, stream>>>(xyzs, img, Kmat, Twc, color, N);
    sigma_kernel<<<1024, 256, 0, stream>>>(xyzs, tab, W1, W2, sigma, N);
}
